// WeatherLSTM_53721450938920
// MI455X (gfx1250) — matrix operand order repaired and checked by reading, not compiled, not run on hardware
//
#include <hip/hip_runtime.h>
#include <hip/hip_bf16.h>

typedef __attribute__((ext_vector_type(16))) _Float16 v16h;
typedef __attribute__((ext_vector_type(8)))  _Float16 v8h;
typedef __attribute__((ext_vector_type(8)))  float    v8f;

#define HID   64
#define SEQ   168
#define DIN   7
#define NB    8192
#define NGATE 256          // 4*HID, gate order i,f,g,o
#define W0K   96           // 64 (h1 recurrent) + 32 (x padded: 7 real + zeros)
#define W1K   128          // 64 (h2 recurrent) + 64 (h1 input)
#define WPB   2            // waves per block
#define BPW   16           // batch rows per wave (WMMA M)
#define BPB   (WPB * BPW)  // 32 batch rows per block

#define LOG2E  1.4426950408889634f

// Branch-free transcendentals: single v_exp_f32 + v_rcp_f32 each.
// Saturate correctly for large |x| (exp2 -> inf -> rcp -> 0).
__device__ __forceinline__ float fast_sigmoid(float x) {
    const float e = __builtin_amdgcn_exp2f(-LOG2E * x);
    return __builtin_amdgcn_rcpf(1.f + e);
}
__device__ __forceinline__ float fast_tanh(float x) {
    const float e = __builtin_amdgcn_exp2f(-2.f * LOG2E * x);
    return fmaf(2.f, __builtin_amdgcn_rcpf(1.f + e), -1.f);
}

// One LSTM layer step over a 16-row batch tile.
// Gate tiles processed in groups of 4 (i,f,g,o slices of one hidden 16-slice).
// B fragments are software double-buffered across the whole layer so every
// v_wmma waits on dscnt<=2 instead of a full LDS drain; the first fragment of
// the next group is in flight while the current group's transcendentals run.
// B-fragment LDS layout: halves [kc][nt][half][np][16] -> frag is 32B contiguous.
template<int NKC>
__device__ __forceinline__ void layer_step(const _Float16* __restrict__ wp,   // per-lane adjusted
                                           const float*    __restrict__ bias,
                                           int np, int hf,
                                           const v16h* A,                     // NKC A-fragments
                                           v8f* c,                            // 4 cell tiles (regs)
                                           _Float16* __restrict__ hout)       // per-wave h buffer
{
    v16h bcur = *(const v16h*)(wp);               // frag(ht=0,kc=0,g=0): nt=0
#pragma unroll
    for (int ht = 0; ht < 4; ++ht) {
        v8f acc[4];
#pragma unroll
        for (int g = 0; g < 4; ++g) {
#pragma unroll
            for (int r = 0; r < 8; ++r) acc[g][r] = 0.f;   // -> inline 0 C operand
        }
#pragma unroll
        for (int kc = 0; kc < NKC; ++kc) {
#pragma unroll
            for (int g = 0; g < 4; ++g) {
                // next fragment coordinates (fully constant-folded under unroll)
                int g2 = g + 1, kc2 = kc, ht2 = ht;
                if (g2 == 4)   { g2 = 0; ++kc2; }
                if (kc2 == NKC){ kc2 = 0; ++ht2; }
                v16h bnext = bcur;
                if (ht2 < 4)
                    bnext = *(const v16h*)(wp + (kc2 * 16 + (g2 * 4 + ht2)) * 512);
                acc[g] = __builtin_amdgcn_wmma_f32_16x16x32_f16(
                    false, A[kc], false, bcur, (short)0, acc[g], false, false);
                bcur = bnext;
            }
        }
        const float bi = bias[(0 * 4 + ht) * 16 + np];
        const float bff = bias[(1 * 4 + ht) * 16 + np];
        const float bg = bias[(2 * 4 + ht) * 16 + np];
        const float bo = bias[(3 * 4 + ht) * 16 + np];
        // elementwise LSTM cell update; C-slot (r,lane) -> (M=r+8*hf, hid=ht*16+np)
#pragma unroll
        for (int r = 0; r < 8; ++r) {
            const float iv = fast_sigmoid(acc[0][r] + bi);
            const float fv = fast_sigmoid(acc[1][r] + bff);
            const float gv = fast_tanh(acc[2][r] + bg);
            const float ov = fast_sigmoid(acc[3][r] + bo);
            const float cn = fv * c[ht][r] + iv * gv;
            c[ht][r] = cn;
            const float hn = ov * fast_tanh(cn);
            hout[(r + 8 * hf) * HID + ht * 16 + np] = (_Float16)hn;
        }
    }
}

__global__ __launch_bounds__(WPB * 32)
void weather_lstm_kernel(const float* __restrict__ x,
                         const float* __restrict__ w_ih0, const float* __restrict__ w_hh0,
                         const float* __restrict__ b_ih0, const float* __restrict__ b_hh0,
                         const float* __restrict__ w_ih1, const float* __restrict__ w_hh1,
                         const float* __restrict__ b_ih1, const float* __restrict__ b_hh1,
                         const float* __restrict__ fc_w,  const float* __restrict__ fc_b,
                         float* __restrict__ out)
{
    extern __shared__ char smem[];
    _Float16* sW0 = (_Float16*)smem;                 // 96*256  f16 (48 KB)
    _Float16* sW1 = sW0 + W0K * NGATE;               // 128*256 f16 (64 KB)
    float*    sB0 = (float*)(sW1 + W1K * NGATE);     // 256 f32
    float*    sB1 = sB0 + NGATE;                     // 256 f32
    float*    sFW = sB1 + NGATE;                     // 4*64 f32
    float*    sFB = sFW + 4 * HID;                   // 4 f32 (+4 pad)
    _Float16* sH  = (_Float16*)(sFB + 8);            // per wave: h1[16*64], h2[16*64]

    const int tid = threadIdx.x;

    // ---- one-time: stage weights into LDS in exact WMMA B-fragment order ----
    // element e = (((kc*16+nt)*2+half)*16+np)*16 + j ; value = W[kc*32+half*16+j][nt*16+np]
    for (int e = tid; e < W0K * NGATE; e += WPB * 32) {
        const int j = e & 15, np_ = (e >> 4) & 15, half = (e >> 8) & 1;
        const int nt = (e >> 9) & 15, kc = e >> 13;
        const int k = kc * 32 + half * 16 + j;
        const int n = nt * 16 + np_;
        float v = 0.f;
        if (k < HID)            v = w_hh0[n * HID + k];
        else if (k < HID + DIN) v = w_ih0[n * DIN + (k - HID)];
        sW0[e] = (_Float16)v;
    }
    for (int e = tid; e < W1K * NGATE; e += WPB * 32) {
        const int j = e & 15, np_ = (e >> 4) & 15, half = (e >> 8) & 1;
        const int nt = (e >> 9) & 15, kc = e >> 13;
        const int k = kc * 32 + half * 16 + j;
        const int n = nt * 16 + np_;
        const float v = (k < HID) ? w_hh1[n * HID + k] : w_ih1[n * HID + (k - HID)];
        sW1[e] = (_Float16)v;
    }
    for (int n = tid; n < NGATE; n += WPB * 32) {
        sB0[n] = b_ih0[n] + b_hh0[n];
        sB1[n] = b_ih1[n] + b_hh1[n];
    }
    for (int i = tid; i < 4 * HID; i += WPB * 32) sFW[i] = fc_w[i];
    if (tid < 4) sFB[tid] = fc_b[tid];
    __syncthreads();

    // ---- per-wave setup: wave owns batch rows [b0, b0+16) ----
    const int wave = tid >> 5;
    const int lane = tid & 31;
    const int hf   = lane >> 4;     // K-half for A/B fragments
    const int np   = lane & 15;     // M (A) / N (B,C) position
    const int b0   = blockIdx.x * BPB + wave * BPW;

    _Float16* h1b = sH + wave * (2 * BPW * HID);
    _Float16* h2b = h1b + BPW * HID;
    for (int i = lane; i < 2 * BPW * HID; i += 32) h1b[i] = (_Float16)0.f;

    const _Float16* w0p = sW0 + (hf * 16 + np) * 16;   // per-lane B-frag base
    const _Float16* w1p = sW1 + (hf * 16 + np) * 16;

    // A-fragment gather from row-major f16 h buffer (two aligned 16B LDS loads)
    auto afrag = [&](const _Float16* hb, int kc) -> v16h {
        const _Float16* p = hb + np * HID + kc * 32 + hf * 16;
        v8h lo = *(const v8h*)p;
        v8h hi = *(const v8h*)(p + 8);
        v16h a;
#pragma unroll
        for (int i = 0; i < 8; ++i) { a[i] = lo[i]; a[i + 8] = hi[i]; }
        return a;
    };

    // x A-fragment for layer0 kchunk 2: lanes 0-15 hold x[M,0..6], rest zero.
    auto xfrag = [&](int t) -> v16h {
        v16h a;
#pragma unroll
        for (int i = 0; i < 16; ++i) a[i] = (_Float16)0.f;
        if (hf == 0) {
            const float* xp = x + (size_t)(b0 + np) * (SEQ * DIN) + (size_t)t * DIN;
#pragma unroll
            for (int d = 0; d < DIN; ++d) a[d] = (_Float16)xp[d];
            if (t + 2 < SEQ) __builtin_prefetch((const void*)(xp + 2 * DIN), 0, 0);
        }
        return a;
    };

    // cell states in registers: [4 hidden tiles][8 C-slots]
    v8f c1[4], c2[4];
#pragma unroll
    for (int i = 0; i < 4; ++i)
#pragma unroll
        for (int r = 0; r < 8; ++r) { c1[i][r] = 0.f; c2[i][r] = 0.f; }

    v16h xa = xfrag(0);
    for (int t = 0; t < SEQ; ++t) {
        const int tn = (t + 1 < SEQ) ? (t + 1) : (SEQ - 1);
        v16h xb = xfrag(tn);                       // double-buffered x load

        v16h A0[3];
        A0[0] = afrag(h1b, 0);                     // h1_{t-1}, K 0..31
        A0[1] = afrag(h1b, 1);                     // h1_{t-1}, K 32..63
        A0[2] = xa;                                // x_t (padded), K 64..95
        layer_step<3>(w0p, sB0, np, hf, A0, c1, h1b);

        v16h A1[4];
        A1[0] = afrag(h2b, 0);                     // h2_{t-1}
        A1[1] = afrag(h2b, 1);
        A1[2] = afrag(h1b, 0);                     // h1_t (just written)
        A1[3] = afrag(h1b, 1);
        layer_step<4>(w1p, sB1, np, hf, A1, c2, h2b);

        xa = xb;
    }

    // ---- FC epilogue: out[b, j] = h2_T . fc_w[j] + fc_b[j] ----
    {
        const int m  = np;
        const int j0 = hf * 2;
#pragma unroll
        for (int jj = 0; jj < 2; ++jj) {
            const int j = j0 + jj;
            float s = sFB[j];
#pragma unroll 8
            for (int k = 0; k < HID; ++k)
                s += (float)h2b[m * HID + k] * sFW[j * HID + k];
            out[(size_t)(b0 + m) * 4 + j] = s;
        }
    }
}

extern "C" void kernel_launch(void* const* d_in, const int* in_sizes, int n_in,
                              void* d_out, int out_size, void* d_ws, size_t ws_size,
                              hipStream_t stream) {
    (void)in_sizes; (void)n_in; (void)out_size; (void)d_ws; (void)ws_size;
    const float* x     = (const float*)d_in[0];
    const float* w_ih0 = (const float*)d_in[1];
    const float* w_hh0 = (const float*)d_in[2];
    const float* b_ih0 = (const float*)d_in[3];
    const float* b_hh0 = (const float*)d_in[4];
    const float* w_ih1 = (const float*)d_in[5];
    const float* w_hh1 = (const float*)d_in[6];
    const float* b_ih1 = (const float*)d_in[7];
    const float* b_hh1 = (const float*)d_in[8];
    const float* fc_w  = (const float*)d_in[9];
    const float* fc_b  = (const float*)d_in[10];

    const size_t lds = (size_t)(W0K + W1K) * NGATE * sizeof(_Float16)    // weights f16
                     + (size_t)(2 * NGATE + 4 * HID + 8) * sizeof(float) // biases + fc
                     + (size_t)WPB * 2 * BPW * HID * sizeof(_Float16);   // h1/h2 per wave

    dim3 grid(NB / BPB);        // 256 blocks
    dim3 block(WPB * 32);       // 2 waves (wave32)
    weather_lstm_kernel<<<grid, block, lds, stream>>>(
        x, w_ih0, w_hh0, b_ih0, b_hh0, w_ih1, w_hh1, b_ih1, b_hh1,
        fc_w, fc_b, (float*)d_out);
}
